// GNN_309237646134
// MI455X (gfx1250) — matrix operand order repaired and checked by reading, not compiled, not run on hardware
//
#include <hip/hip_runtime.h>
#include <hip/hip_bf16.h>

// ---------------------------------------------------------------------------
// GNN forward for MI455X (gfx1250, wave32, WMMA).
// - Dense projections: v_wmma_f32_16x16x32_f16 (f16 in, f32 acc), 128x64 WG
//   tile, B weight panel resident in LDS via global_load_async_to_lds_b128.
// - Edge aggregation: global_atomic_add_f32 (unsafeAtomicAdd), feature-
//   parallel coalesced.
// ---------------------------------------------------------------------------

typedef __attribute__((ext_vector_type(16))) _Float16 v16h;
typedef __attribute__((ext_vector_type(8)))  _Float16 v8h;
typedef __attribute__((ext_vector_type(4)))  _Float16 v4h;
typedef __attribute__((ext_vector_type(8)))  float    v8f;

#define TM 128
#define TN 64
#define TK 32
#define KPAD_MAX 256   // max padded K (Bs panel sized for this)

__device__ __forceinline__ float lrelu(float v) { return v > 0.f ? v : 0.01f * v; }

// ----------------------------- small kernels -------------------------------
__global__ void k_fill(float* p, float v, int n) {
    int i = blockIdx.x * blockDim.x + threadIdx.x;
    if (i < n) p[i] = v;
}

__global__ void k_count_deg(const int* __restrict__ dst, float* __restrict__ deg, int E) {
    int e = blockIdx.x * blockDim.x + threadIdx.x;
    if (e < E) unsafeAtomicAdd(&deg[dst[e]], 1.0f);
}

__global__ void k_rsqrt_inplace(float* p, int n) {
    int i = blockIdx.x * blockDim.x + threadIdx.x;
    if (i < n) p[i] = rsqrtf(p[i]);
}

__global__ void k_zero(float* p, int n) {
    int i = blockIdx.x * blockDim.x + threadIdx.x;
    if (i < n) p[i] = 0.0f;
}

// --------------- weight fp32 [K,Nc] -> fp16 transposed [Nc,Kpad] -----------
// K-contiguous rows so the GEMM B panel is async-DMA friendly (16B chunks).
__global__ void k_cvt_wT(const float* __restrict__ W, _Float16* __restrict__ Wt,
                         int K, int Kpad, int Nc) {
    int i = blockIdx.x * blockDim.x + threadIdx.x;   // i = n*Kpad + k
    if (i >= Nc * Kpad) return;
    int n = i / Kpad;
    int k = i - n * Kpad;
    Wt[i] = (k < K) ? (_Float16)W[(size_t)k * Nc + n] : (_Float16)0.f;
}

// ------------------------------ WMMA GEMM ----------------------------------
// C[M,Nc] = act( A_f32[M,K] * Wt_f16[Nc,Kpad]^T (+ bias) )
// 256 threads = 8 waves; WG tile 128x64; wave tile 32x32 = 2x2 WMMA frags.
// B panel (64 x Kpad f16, <=32KB) staged ONCE per WG with async LDS DMA.
__global__ __launch_bounds__(256) void k_gemm_wmma(
    const float* __restrict__ A, const _Float16* __restrict__ Wt,
    const float* __restrict__ bias, float* __restrict__ C,
    int M, int K, int Kpad, int Nc, int add_bias, int do_lrelu)
{
    __shared__ __align__(64) _Float16 As[TM * TK];        // 8 KB  [m][k]
    __shared__ __align__(64) _Float16 Bs[TN * KPAD_MAX];  // 32 KB [n][k], k-contig

    const int tid   = threadIdx.x;
    const int lane  = tid & 31;
    const int wave  = tid >> 5;
    const int waveM = wave & 3;   // 4 waves along M -> 128 rows
    const int waveN = wave >> 2;  // 2 waves along N -> 64 cols
    const int m0 = blockIdx.y * TM;
    const int n0 = blockIdx.x * TN;

    // ---- stage entire B panel: rows n0..n0+63 of Wt, Kpad halves each ----
    {
        const int chunks = (TN * Kpad) >> 3;        // 16B chunks (8 halves)
        const int cpr    = Kpad >> 3;               // chunks per row
        for (int c = tid; c < chunks; c += 256) {
            int n  = c / cpr;
            int kc = c - n * cpr;
            const _Float16* gp = Wt + (size_t)(n0 + n) * Kpad + (kc << 3);
            unsigned ldsoff = (unsigned)(uintptr_t)(&Bs[n * Kpad + (kc << 3)]);
            asm volatile("global_load_async_to_lds_b128 %0, %1, off"
                         :: "v"(ldsoff), "v"(gp) : "memory");
        }
        asm volatile("s_wait_asynccnt 0x0" ::: "memory");
    }
    __syncthreads();   // B panel resident for all waves

    v8f acc[2][2] = {};

    const int ar = tid >> 3;          // 0..31 (A tile sub-row)
    const int ac = (tid & 7) * 4;     // 0..28 (A tile col quad)
    const bool mfull = (m0 + TM <= M);

    const int mlo = lane & 15;
    const int kbA = (lane >> 4) * 16;
    const int kbB = (lane >> 4) * 16;

    for (int k0 = 0; k0 < Kpad; k0 += TK) {
        // --- stage A tile (fp32 -> fp16) ----------------------------------
        if (mfull && (k0 + TK <= K)) {
            // interior: branch-free float4 load + packed 8B LDS store
            #pragma unroll
            for (int rr = 0; rr < 4; ++rr) {
                int row = ar + rr * 32;
                float4 v = *(const float4*)(A + (size_t)(m0 + row) * K + k0 + ac);
                v4h h = { (_Float16)v.x, (_Float16)v.y, (_Float16)v.z, (_Float16)v.w };
                *(v4h*)&As[row * TK + ac] = h;
            }
        } else {
            #pragma unroll
            for (int rr = 0; rr < 4; ++rr) {
                int row = ar + rr * 32;
                int gm  = m0 + row;
                float f0 = 0.f, f1 = 0.f, f2 = 0.f, f3 = 0.f;
                if (gm < M) {
                    int k = k0 + ac;
                    const float* p = A + (size_t)gm * K + k;
                    if (k + 4 <= K) {
                        float4 v = *(const float4*)p;
                        f0 = v.x; f1 = v.y; f2 = v.z; f3 = v.w;
                    } else {
                        if (k + 0 < K) f0 = p[0];
                        if (k + 1 < K) f1 = p[1];
                        if (k + 2 < K) f2 = p[2];
                        if (k + 3 < K) f3 = p[3];
                    }
                }
                v4h h = { (_Float16)f0, (_Float16)f1, (_Float16)f2, (_Float16)f3 };
                *(v4h*)&As[row * TK + ac] = h;
            }
        }
        __syncthreads();

        // --- WMMA, contiguous-k fragments (same order for A and B) ------------------
        // A 16x32 f16: lanes 0-15 M=lane, K {0..15};
        //              lanes 16-31 M=lane-16, K {16..31}.
        // B 32x16 f16 from k-contiguous LDS rows: one 32B load per frag.
        #pragma unroll
        for (int i = 0; i < 2; ++i) {
            int mrow = waveM * 32 + 16 * i + mlo;
            v8h alo = *(const v8h*)&As[mrow * TK + kbA];
            v8h ahi = *(const v8h*)&As[mrow * TK + 8 + kbA];
            v16h a  = __builtin_shufflevector(alo, ahi,
                        0,1,2,3,4,5,6,7,8,9,10,11,12,13,14,15);
            #pragma unroll
            for (int j = 0; j < 2; ++j) {
                int ncol = waveN * 32 + 16 * j + mlo;
                v16h b = *(const v16h*)&Bs[ncol * Kpad + k0 + kbB];
                acc[i][j] = __builtin_amdgcn_wmma_f32_16x16x32_f16(
                    false, a, false, b, (short)0, acc[i][j], false, false);
            }
        }
        __syncthreads();
    }

    // --- epilogue: C frag layout VGPR r -> M = r + 8*(lane>>4), N = lane&15
    const int mbase = m0 + waveM * 32;
    const int nbase = n0 + waveN * 32;
    #pragma unroll
    for (int i = 0; i < 2; ++i) {
        #pragma unroll
        for (int j = 0; j < 2; ++j) {
            int nn = nbase + 16 * j + (lane & 15);
            float bval = add_bias ? bias[nn] : 0.0f;
            #pragma unroll
            for (int r = 0; r < 8; ++r) {
                int mm = mbase + 16 * i + r + 8 * (lane >> 4);
                if (mfull || mm < M) {
                    float v = acc[i][j][r] + bval;
                    if (do_lrelu) v = lrelu(v);
                    C[(size_t)mm * Nc + nn] = v;
                }
            }
        }
    }
}

// --------------------------- GCN aggregation -------------------------------
__global__ void k_agg_init(const float* __restrict__ xw, const float* __restrict__ dinv,
                           const float* __restrict__ bias, float* __restrict__ out,
                           int Nn, int F) {
    size_t i = (size_t)blockIdx.x * blockDim.x + threadIdx.x;
    if (i >= (size_t)Nn * F) return;
    int n = (int)(i / F);
    int f = (int)(i - (size_t)n * F);
    float di = dinv[n];
    out[i] = xw[i] * di * di + bias[f];
}

__global__ __launch_bounds__(256) void k_agg_edges(
    const float* __restrict__ xw, const int* __restrict__ src,
    const int* __restrict__ dst, const float* __restrict__ dinv,
    float* __restrict__ out, int E, int F)
{
    int epb = 256 / F;
    int le  = threadIdx.x / F;
    int f   = threadIdx.x - le * F;
    int e   = blockIdx.x * epb + le;
    if (e >= E) return;
    int s = src[e], d = dst[e];
    float w = dinv[s] * dinv[d];
    unsafeAtomicAdd(&out[(size_t)d * F + f], xw[(size_t)s * F + f] * w);
}

// -------------------- LayerNorm + leakyReLU (+ L2 norm) --------------------
__global__ void k_ln_lrelu(const float* __restrict__ in, const float* __restrict__ g,
                           const float* __restrict__ bta, float* __restrict__ out,
                           int Nn, int F, int do_l2)
{
    int node = blockIdx.x * (blockDim.x >> 5) + (threadIdx.x >> 5);
    int lane = threadIdx.x & 31;
    if (node >= Nn) return;
    const float* row = in + (size_t)node * F;
    float v[8];
    int nper = F >> 5;
    float s = 0.f;
    for (int i = 0; i < nper; ++i) { v[i] = row[lane + 32 * i]; s += v[i]; }
    #pragma unroll
    for (int off = 16; off; off >>= 1) s += __shfl_xor(s, off);
    float mu = s / F;
    float q = 0.f;
    for (int i = 0; i < nper; ++i) { float d = v[i] - mu; q += d * d; }
    #pragma unroll
    for (int off = 16; off; off >>= 1) q += __shfl_xor(q, off);
    float rstd = rsqrtf(q / F + 1e-5f);
    float ss = 0.f;
    for (int i = 0; i < nper; ++i) {
        int f = lane + 32 * i;
        float y = (v[i] - mu) * rstd * g[f] + bta[f];
        y = lrelu(y);
        v[i] = y;
        ss += y * y;
    }
    if (do_l2) {
        #pragma unroll
        for (int off = 16; off; off >>= 1) ss += __shfl_xor(ss, off);
        float scale = 1.0f / fmaxf(sqrtf(ss), 1e-12f);
        for (int i = 0; i < nper; ++i) v[i] *= scale;
    }
    float* orow = out + (size_t)node * F;
    for (int i = 0; i < nper; ++i) orow[lane + 32 * i] = v[i];
}

// ------------------------------ pooling ------------------------------------
__global__ void k_pool(const float* __restrict__ ha, const int* __restrict__ batch,
                       float* __restrict__ hg, int Nn) {
    size_t i = (size_t)blockIdx.x * blockDim.x + threadIdx.x;
    if (i >= (size_t)Nn * 128) return;
    int n = (int)(i >> 7), f = (int)(i & 127);
    unsafeAtomicAdd(&hg[batch[n] * 128 + f], ha[i]);
}

// ------------------------------ MLP head -----------------------------------
__global__ __launch_bounds__(64) void k_head(
    const float* __restrict__ hg, const float* __restrict__ W1,
    const float* __restrict__ b1, const float* __restrict__ W2,
    const float* __restrict__ b2, float* __restrict__ out, int Gn)
{
    int g = blockIdx.x;
    int t = threadIdx.x;
    if (g >= Gn) return;
    __shared__ float sh[64];
    const float* row = hg + (size_t)g * 128;
    float acc = b1[t];
    for (int k = 0; k < 128; ++k) acc += row[k] * W1[k * 64 + t];
    sh[t] = lrelu(acc) * W2[t];
    __syncthreads();
    if (t == 0) {
        float s = b2[0];
        for (int k = 0; k < 64; ++k) s += sh[k];
        out[g] = s;
    }
}

// ---------------------------------------------------------------------------
extern "C" void kernel_launch(void* const* d_in, const int* in_sizes, int n_in,
                              void* d_out, int out_size, void* d_ws, size_t ws_size,
                              hipStream_t stream) {
    const float* x     = (const float*)d_in[0];
    const int*   edge  = (const int*)d_in[1];
    const int*   batch = (const int*)d_in[2];
    const float* nfc_W = (const float*)d_in[3];
    const float* nfc_b = (const float*)d_in[4];
    const float* gn1_g = (const float*)d_in[5];
    const float* gn1_b = (const float*)d_in[6];
    const float* gc1_W = (const float*)d_in[7];
    const float* gc1_b = (const float*)d_in[8];
    const float* gn2_g = (const float*)d_in[9];
    const float* gn2_b = (const float*)d_in[10];
    const float* gc2_W = (const float*)d_in[11];
    const float* gc2_b = (const float*)d_in[12];
    const float* fc1_W = (const float*)d_in[13];
    const float* fc1_b = (const float*)d_in[14];
    const float* fc2_W = (const float*)d_in[15];
    const float* fc2_b = (const float*)d_in[16];

    const int N  = in_sizes[2];
    const int E  = in_sizes[1] / 2;
    const int K1 = in_sizes[0] / N;        // 92
    const int K1p = ((K1 + 31) / 32) * 32; // 96
    const int DH = 256, DE = 128;
    const int G  = out_size - N * DE;      // 500

    const int* srcp = edge;
    const int* dstp = edge + E;

    char* ws = (char*)d_ws;
    size_t off = 0;
    auto carve = [&](size_t bytes) -> void* {
        void* p = ws + off;
        off = (off + bytes + 255) & ~(size_t)255;
        return p;
    };
    float*     deg  = (float*)carve((size_t)N * 4);            // -> dinv in place
    _Float16*  Wt0  = (_Float16*)carve((size_t)DH * K1p * 2);  // [256,96]
    _Float16*  Wt1  = (_Float16*)carve((size_t)DH * DH * 2);   // [256,256]
    _Float16*  Wt2  = (_Float16*)carve((size_t)DE * DH * 2);   // [128,256]
    float*     bufA = (float*)carve((size_t)N * DH * 4);
    float*     bufB = (float*)carve((size_t)N * DH * 4);
    float*     bufC = (float*)carve((size_t)N * DH * 4);
    float*     hg   = (float*)carve((size_t)G * DE * 4);

    float* out_head = (float*)d_out;
    float* ha_out   = (float*)d_out + G;

    // 1) degrees -> dinv
    k_fill<<<(N + 255) / 256, 256, 0, stream>>>(deg, 1.0f, N);
    k_count_deg<<<(E + 255) / 256, 256, 0, stream>>>(dstp, deg, E);
    k_rsqrt_inplace<<<(N + 255) / 256, 256, 0, stream>>>(deg, N);

    // 2) weights -> f16, transposed [Nc, Kpad] (K-contiguous for async DMA)
    k_cvt_wT<<<(DH * K1p + 255) / 256, 256, 0, stream>>>(nfc_W, Wt0, K1, K1p, DH);
    k_cvt_wT<<<(DH * DH + 255) / 256, 256, 0, stream>>>(gc1_W, Wt1, DH, DH, DH);
    k_cvt_wT<<<(DE * DH + 255) / 256, 256, 0, stream>>>(gc2_W, Wt2, DH, DH, DE);

    const int mt = (N + TM - 1) / TM;

    // 3) ha = lrelu(x @ nfc_W + b)
    k_gemm_wmma<<<dim3(DH / TN, mt), 256, 0, stream>>>(
        x, Wt0, nfc_b, bufA, N, K1, K1p, DH, 1, 1);

    // 4) conv1
    k_gemm_wmma<<<dim3(DH / TN, mt), 256, 0, stream>>>(
        bufA, Wt1, nullptr, bufB, N, DH, DH, DH, 0, 0);
    k_agg_init<<<(int)(((size_t)N * DH + 255) / 256), 256, 0, stream>>>(
        bufB, deg, gc1_b, bufC, N, DH);
    k_agg_edges<<<E, 256, 0, stream>>>(bufB, srcp, dstp, deg, bufC, E, DH);
    k_ln_lrelu<<<(N + 7) / 8, 256, 0, stream>>>(bufC, gn1_g, gn1_b, bufA, N, DH, 0);

    // 5) conv2 (+ L2 normalize -> ha output region)
    k_gemm_wmma<<<dim3(DE / TN, mt), 256, 0, stream>>>(
        bufA, Wt2, nullptr, bufB, N, DH, DH, DE, 0, 0);
    k_agg_init<<<(int)(((size_t)N * DE + 255) / 256), 256, 0, stream>>>(
        bufB, deg, gc2_b, bufC, N, DE);
    k_agg_edges<<<(E + 1) / 2, 256, 0, stream>>>(bufB, srcp, dstp, deg, bufC, E, DE);
    k_ln_lrelu<<<(N + 7) / 8, 256, 0, stream>>>(bufC, gn2_g, gn2_b, ha_out, N, DE, 1);

    // 6) pool + head
    k_zero<<<(G * DE + 255) / 256, 256, 0, stream>>>(hg, G * DE);
    k_pool<<<(int)(((size_t)N * DE + 255) / 256), 256, 0, stream>>>(ha_out, batch, hg, N);
    k_head<<<G, 64, 0, stream>>>(hg, fc1_W, fc1_b, fc2_W, fc2_b, out_head, G);
}
